// CausalAttention_38972533244086
// MI455X (gfx1250) — matrix operand order repaired and checked by reading, not compiled, not run on hardware
//
#include <hip/hip_runtime.h>
#include <cstdint>

// ---------------------------------------------------------------------------
// CDNA5 (gfx1250, wave32) causal attention via v_wmma_f32_16x16x32_bf16,
// with async global->LDS tile copies (ASYNCcnt) and double-buffered LDS.
// ---------------------------------------------------------------------------

typedef __bf16 bf16_t;
typedef __attribute__((ext_vector_type(16))) __bf16 v16bf;
typedef __attribute__((ext_vector_type(8)))  float  v8f;

__device__ __forceinline__ bf16_t f32_to_bf16(float f) {
    uint32_t u = __builtin_bit_cast(uint32_t, f);
    if ((u & 0x7fffffffu) > 0x7f800000u) {
        u |= 0x00400000u;               // quiet the NaN, then truncate
    } else {
        u += 0x7fffu + ((u >> 16) & 1u); // round-to-nearest-even
    }
    uint16_t h = (uint16_t)(u >> 16);
    return __builtin_bit_cast(bf16_t, h);
}

// Async 16-byte global -> LDS copy (per lane), tracked by ASYNCcnt.
__device__ __forceinline__ void async_ld16(uint32_t lds_addr, const void* gaddr) {
    asm volatile("global_load_async_to_lds_b128 %0, %1, off"
                 :: "v"(lds_addr), "v"(gaddr) : "memory");
}

// ---------------------------------------------------------------------------
// Elementwise fp32 -> bf16 (for x), 4 elements per thread.
// ---------------------------------------------------------------------------
__global__ __launch_bounds__(256) void convert_f32_to_bf16(
    const float* __restrict__ src, bf16_t* __restrict__ dst, int n)
{
    int i = (blockIdx.x * 256 + threadIdx.x) * 4;
    if (i + 3 < n) {
        float4 f = *reinterpret_cast<const float4*>(src + i);
        dst[i + 0] = f32_to_bf16(f.x);
        dst[i + 1] = f32_to_bf16(f.y);
        dst[i + 2] = f32_to_bf16(f.z);
        dst[i + 3] = f32_to_bf16(f.w);
    }
}

// ---------------------------------------------------------------------------
// Transpose-convert 1024x1024 fp32 W[k][n] -> bf16 Wt[n][k]; grid.z picks W.
// ---------------------------------------------------------------------------
__global__ __launch_bounds__(256) void transpose_w_to_bf16(
    const float* __restrict__ W0, const float* __restrict__ W1,
    const float* __restrict__ W2, bf16_t* __restrict__ Wt)
{
    const float* W = (blockIdx.z == 0) ? W0 : ((blockIdx.z == 1) ? W1 : W2);
    bf16_t* T = Wt + (size_t)blockIdx.z * 1024 * 1024;
    __shared__ float tile[32][33];
    const int k0 = blockIdx.y * 32;
    const int n0 = blockIdx.x * 32;
    const int tx = threadIdx.x, ty = threadIdx.y;
#pragma unroll
    for (int i = 0; i < 32; i += 8)
        tile[ty + i][tx] = W[(long)(k0 + ty + i) * 1024 + (n0 + tx)];
    __syncthreads();
#pragma unroll
    for (int i = 0; i < 32; i += 8)
        T[(long)(n0 + ty + i) * 1024 + (k0 + tx)] = f32_to_bf16(tile[tx][ty + i]);
}

// ---------------------------------------------------------------------------
// bf16 GEMM via WMMA:  C[m][n] = sum_k A[m][k] * Bt[n][k]
//   - 128 threads = 4 waves; workgroup tile 128(M) x 64(N)
//   - wave w owns rows [32w,32w+32): 2 A-fragments x 4 B-fragments = 8 WMMAs
//   - K-steps of 32; double-buffered LDS filled by async global->LDS copies
//     (6 x b128 per thread per tile), pipelined one K-step ahead;
//     steady-state loop is branch-free (last iteration peeled)
//   - mode: 0 = fp32 store, 1 = bf16 store, 2 = bf16 transposed store
//   - causal: 0 none; 1 skip workgroups fully above diagonal (scores GEMM);
//             2 truncate K loop at the diagonal (P @ V GEMM)
// All of M, N, K are multiples of the tile sizes here; no edge handling.
// ---------------------------------------------------------------------------
__global__ __launch_bounds__(128) void gemm_bf16_wmma(
    const bf16_t* __restrict__ A,  long aBatch, int lda,
    const bf16_t* __restrict__ Bt, long bBatch, int ldb,
    void* __restrict__ Cout,       long cBatch, int ldc,
    int Kdim, int mode, float scale, int causal)
{
    const int m0 = blockIdx.y * 128;
    const int n0 = blockIdx.x * 64;
    if (causal == 1 && n0 > m0 + 127) return;   // fully masked score tile
    int kEnd = Kdim;
    if (causal == 2) {
        int lim = (blockIdx.y + 1) * 128;       // keys <= max query row in tile
        kEnd = (lim < Kdim) ? lim : Kdim;
    }

    A  += (long)blockIdx.z * aBatch;
    Bt += (long)blockIdx.z * bBatch;

    __shared__ __align__(16) bf16_t As[2][128][40];  // double-buffered A tile
    __shared__ __align__(16) bf16_t Bs[2][64][40];   // double-buffered B tile

    const int tid  = threadIdx.x;
    const int wave = tid >> 5;
    const int lane = tid & 31;
    const int g    = lane >> 4;    // half-wave select (K-group)
    const int mn   = lane & 15;    // M index (A) / N index (B,C,D)

    // Issue one K-step tile's async copies: 4 A chunks + 2 B chunks / thread.
    auto issue_tile = [&](int kb, int buf) {
#pragma unroll
        for (int i = 0; i < 4; ++i) {
            int c  = tid + 128 * i;          // 0..511
            int r  = c >> 2;                 // 0..127
            int co = (c & 3) << 3;           // 0,8,16,24 (bf16 elems)
            async_ld16((uint32_t)(uintptr_t)&As[buf][r][co],
                       A + (long)(m0 + r) * lda + kb + co);
        }
#pragma unroll
        for (int i = 0; i < 2; ++i) {
            int c  = tid + 128 * i;          // 0..255
            int r  = c >> 2;                 // 0..63
            int co = (c & 3) << 3;
            async_ld16((uint32_t)(uintptr_t)&Bs[buf][r][co],
                       Bt + (long)(n0 + r) * ldb + kb + co);
        }
    };

    const v8f vzero = {0.f, 0.f, 0.f, 0.f, 0.f, 0.f, 0.f, 0.f};
    v8f acc[2][4];
#pragma unroll
    for (int s = 0; s < 2; ++s)
#pragma unroll
        for (int t = 0; t < 4; ++t) acc[s][t] = vzero;

    // One K-step of WMMA on LDS buffer `buf`.
    union frag16 { uint4 u[2]; v16bf v; };
    auto compute_tile = [&](int buf) {
        frag16 af[2];
#pragma unroll
        for (int s = 0; s < 2; ++s) {
            int r = wave * 32 + s * 16 + mn;                 // M = lane%16
            af[s].u[0] = *reinterpret_cast<const uint4*>(&As[buf][r][g * 8]);
            af[s].u[1] = *reinterpret_cast<const uint4*>(&As[buf][r][16 + g * 8]);
        }
#pragma unroll
        for (int t = 0; t < 4; ++t) {
            // B fragment (32x16): lane n = lane%16, K = g*8..g*8+7 and 16+g*8..16+g*8+7
            frag16 bfr;
            int r = t * 16 + mn;
            bfr.u[0] = *reinterpret_cast<const uint4*>(&Bs[buf][r][g * 8]);
            bfr.u[1] = *reinterpret_cast<const uint4*>(&Bs[buf][r][16 + g * 8]);
#pragma unroll
            for (int s = 0; s < 2; ++s)
                acc[s][t] = __builtin_amdgcn_wmma_f32_16x16x32_bf16(
                    false, af[s].v, false, bfr.v, (short)0, acc[s][t], false, false);
        }
    };

    const int nk = kEnd >> 5;   // >= 1 always here
    issue_tile(0, 0);

    int it = 0;
    for (; it < nk - 1; ++it) {                  // steady state: branch-free
        issue_tile((it + 1) << 5, (it & 1) ^ 1);
        asm volatile("s_wait_asynccnt 6" ::: "memory");  // tile `it` resident
        __syncthreads();
        compute_tile(it & 1);
        __syncthreads();       // reads done before this buf is refilled
    }
    asm volatile("s_wait_asynccnt 0" ::: "memory");      // last tile resident
    __syncthreads();
    compute_tile(it & 1);

    // --- store: C/D layout: lane -> N = lane%16, VGPR v -> M = v + 8*(lane/16)
    if (mode == 0) {
        float* C = (float*)Cout + (long)blockIdx.z * cBatch;
#pragma unroll
        for (int s = 0; s < 2; ++s)
#pragma unroll
            for (int t = 0; t < 4; ++t)
#pragma unroll
                for (int v = 0; v < 8; ++v) {
                    int row = m0 + wave * 32 + s * 16 + v + 8 * g;
                    int col = n0 + t * 16 + mn;
                    C[(long)row * ldc + col] = acc[s][t][v] * scale;
                }
    } else if (mode == 1) {
        bf16_t* C = (bf16_t*)Cout + (long)blockIdx.z * cBatch;
#pragma unroll
        for (int s = 0; s < 2; ++s)
#pragma unroll
            for (int t = 0; t < 4; ++t)
#pragma unroll
                for (int v = 0; v < 8; ++v) {
                    int row = m0 + wave * 32 + s * 16 + v + 8 * g;
                    int col = n0 + t * 16 + mn;
                    C[(long)row * ldc + col] = f32_to_bf16(acc[s][t][v] * scale);
                }
    } else {  // bf16 transposed store: C[col][row]
        bf16_t* C = (bf16_t*)Cout + (long)blockIdx.z * cBatch;
#pragma unroll
        for (int s = 0; s < 2; ++s)
#pragma unroll
            for (int t = 0; t < 4; ++t)
#pragma unroll
                for (int v = 0; v < 8; ++v) {
                    int row = m0 + wave * 32 + s * 16 + v + 8 * g;
                    int col = n0 + t * 16 + mn;
                    C[(long)col * ldc + row] = f32_to_bf16(acc[s][t][v] * scale);
                }
    }
}

// ---------------------------------------------------------------------------
// Row softmax with causal mask. One 256-thread block per (b,q) row.
// Reads fp32 scores, writes normalized bf16 P (zeros above the diagonal).
// ---------------------------------------------------------------------------
__global__ __launch_bounds__(256) void softmax_causal(
    const float* __restrict__ S, bf16_t* __restrict__ P)
{
    const long row = blockIdx.x;           // 0 .. B*S-1
    const int  q   = (int)(row & 2047);
    const float* s = S + row * 2048;
    bf16_t*      p = P + row * 2048;
    __shared__ float red[256];
    const int tid = threadIdx.x;

    float m = -3.4e38f;
    for (int k = tid; k <= q; k += 256) m = fmaxf(m, s[k]);
    red[tid] = m; __syncthreads();
    for (int off = 128; off > 0; off >>= 1) {
        if (tid < off) red[tid] = fmaxf(red[tid], red[tid + off]);
        __syncthreads();
    }
    const float rowmax = red[0];
    __syncthreads();

    float sum = 0.f;
    for (int k = tid; k <= q; k += 256) sum += __expf(s[k] - rowmax);
    red[tid] = sum; __syncthreads();
    for (int off = 128; off > 0; off >>= 1) {
        if (tid < off) red[tid] += red[tid + off];
        __syncthreads();
    }
    const float inv = 1.0f / red[0];

    const bf16_t zero = f32_to_bf16(0.f);
    for (int k = tid; k < 2048; k += 256)
        p[k] = (k <= q) ? f32_to_bf16(__expf(s[k] - rowmax) * inv) : zero;
}

// ---------------------------------------------------------------------------
// Host-side orchestration.
//   inputs: x [4,2048,1024] f32; W_query/W_key/W_value [1024,1024] f32
//   output: [4,2048,1024] f32
// Workspace layout (bytes):
//   xb   bf16  16 MB @ 0
//   Wt   bf16   6 MB @ 16 MB   (Wq^T, Wk^T, Wv^T, each 1024x1024)
//   Qb   bf16  16 MB @ 22 MB   (pre-scaled by 1/sqrt(d))
//   Kb   bf16  16 MB @ 38 MB
//   Vt   bf16  16 MB @ 54 MB   (per-batch transposed: [b][feat][seq])
//   S    f32   64 MB @ 70 MB
//   P    bf16  32 MB @ 134 MB
// ---------------------------------------------------------------------------
extern "C" void kernel_launch(void* const* d_in, const int* in_sizes, int n_in,
                              void* d_out, int out_size, void* d_ws, size_t ws_size,
                              hipStream_t stream)
{
    (void)in_sizes; (void)n_in; (void)out_size; (void)ws_size;
    const float* x  = (const float*)d_in[0];
    const float* Wq = (const float*)d_in[1];
    const float* Wk = (const float*)d_in[2];
    const float* Wv = (const float*)d_in[3];
    float* out = (float*)d_out;

    char* ws = (char*)d_ws;
    bf16_t* xb = (bf16_t*)(ws);
    bf16_t* Wt = (bf16_t*)(ws + (16ull << 20));
    bf16_t* Qb = (bf16_t*)(ws + (22ull << 20));
    bf16_t* Kb = (bf16_t*)(ws + (38ull << 20));
    bf16_t* Vt = (bf16_t*)(ws + (54ull << 20));
    float*  Sb = (float*) (ws + (70ull << 20));
    bf16_t* Pb = (bf16_t*)(ws + (134ull << 20));

    const int  B = 4, S = 2048, D = 1024;
    const long MK = (long)S * D;          // per-batch Q/K elements
    const long SS = (long)S * S;          // per-batch score elements
    const float qscale = 0.03125f;        // 1/sqrt(1024)

    // 1. x -> bf16
    convert_f32_to_bf16<<<(B * S * D) / 1024, 256, 0, stream>>>(x, xb, B * S * D);
    // 2. W -> Wt (bf16, transposed), all three
    transpose_w_to_bf16<<<dim3(32, 32, 3), dim3(32, 8), 0, stream>>>(Wq, Wk, Wv, Wt);

    // 3. Q = (x Wq) * 1/sqrt(d)   [bf16, row-major]
    gemm_bf16_wmma<<<dim3(D / 64, S / 128, B), 128, 0, stream>>>(
        xb, MK, D,  Wt + 0 * D * D, 0, D,  Qb, MK, D,  D, 1, qscale, 0);
    // 4. K = x Wk                 [bf16, row-major]
    gemm_bf16_wmma<<<dim3(D / 64, S / 128, B), 128, 0, stream>>>(
        xb, MK, D,  Wt + 1 * (long)D * D, 0, D,  Kb, MK, D,  D, 1, 1.0f, 0);
    // 5. V^T = (x Wv)^T           [bf16, transposed per batch: [feat][seq]]
    gemm_bf16_wmma<<<dim3(D / 64, S / 128, B), 128, 0, stream>>>(
        xb, MK, D,  Wt + 2 * (long)D * D, 0, D,  Vt, MK, S,  D, 2, 1.0f, 0);

    // 6. Scores = Q K^T  [f32], skipping tiles fully above the diagonal
    gemm_bf16_wmma<<<dim3(S / 64, S / 128, B), 128, 0, stream>>>(
        Qb, MK, D,  Kb, MK, D,  Sb, SS, S,  D, 0, 1.0f, 1);

    // 7. Causal softmax -> P (bf16, zero above diagonal)
    softmax_causal<<<B * S, 256, 0, stream>>>(Sb, Pb);

    // 8. Out = P V  [f32], K-loop truncated at the diagonal
    gemm_bf16_wmma<<<dim3(D / 64, S / 128, B), 128, 0, stream>>>(
        Pb, SS, S,  Vt, MK, S,  out, MK, D,  S, 0, 1.0f, 2);
}
